// Window_Attention_Intent_Decoder_52192442581780
// MI455X (gfx1250) — matrix operand order repaired and checked by reading, not compiled, not run on hardware
//
#include <hip/hip_runtime.h>
#include <hip/hip_fp16.h>
#include <hip/hip_bf16.h>

#define DEVINL __device__ __forceinline__

typedef __attribute__((ext_vector_type(16))) _Float16 v16h;
typedef __attribute__((ext_vector_type(8)))  _Float16 v8h;
typedef __attribute__((ext_vector_type(8)))  float    v8f;

// ---------------- problem constants ----------------
constexpr int   B_   = 8;
constexpr int   S_   = 2048;
constexpr int   E_   = 256;
constexpr int   WS_  = 5;
constexpr int   H_   = 4;
constexpr int   HD_  = 64;
constexpr int   FF_  = 2048;
constexpr int   NI_  = 64;
constexpr int   W_   = S_ - WS_ + 1;     // 2044
constexpr int   NWIN = B_ * W_;          // 16352 windows
constexpr int   NROW = NWIN * WS_;       // 81760 encoder rows
constexpr int   NTOK = B_ * S_;          // 16384 tokens
constexpr float EPS_ = 1e-5f;

// ---------------- WMMA helpers (CDNA5, wave32) ----------------
DEVINL v8f wmma16x16x32(v16h a, v16h b, v8f c) {
  return __builtin_amdgcn_wmma_f32_16x16x32_f16(
      /*neg_a=*/false, a, /*neg_b=*/false, b,
      /*c_mod=*/(short)0, c, /*reuse_a=*/false, /*reuse_b=*/false);
}

// A fragment: rows [m0,m0+16), K [k0,k0+32) from row-major MxK f16.
// Layout: lane m = lane&15; lane[4] selects K+0..15 vs K+16..31 (matches B frag).
DEVINL v16h load_a_frag(const _Float16* __restrict__ A, int ldk, int m0, int k0,
                        int maxRow) {
  int lane = threadIdx.x & 31;
  int row  = m0 + (lane & 15);
  if (row > maxRow) row = maxRow;
  const _Float16* p = A + (size_t)row * ldk + k0 + ((lane >> 4) << 4);
  v8h lo = *(const v8h*)p;
  v8h hi = *(const v8h*)(p + 8);
  v16h f;
#pragma unroll
  for (int i = 0; i < 8; ++i) { f[i] = lo[i]; f[i + 8] = hi[i]; }
  return f;
}

// B fragment from weight matrix W (N x K row-major), computing X @ W^T:
// B[k][n] = W[n][k]. Lane holds column n0+(lane&15); lanes 16-31 take K+16..31.
DEVINL v16h load_b_frag(const _Float16* __restrict__ Wm, int ldk, int n0, int k0) {
  int lane = threadIdx.x & 31;
  int col  = n0 + (lane & 15);
  const _Float16* p = Wm + (size_t)col * ldk + k0 + ((lane >> 4) << 4);
  return *(const v16h*)p;  // 16 contiguous halves (two b128 loads)
}

// C/D layout: col = n0+(lane&15); rows = m0 + 8*lane[4] + r.
DEVINL void store_c_f32(float* __restrict__ C, int ldc, int M, int m0, int n0,
                        v8f acc, const float* __restrict__ bias, int relu) {
  int lane = threadIdx.x & 31;
  int col  = n0 + (lane & 15);
  int rb   = m0 + ((lane >> 4) << 3);
  float bv = bias[col];
#pragma unroll
  for (int r = 0; r < 8; ++r) {
    int row = rb + r;
    if (row < M) {
      float v = acc[r] + bv;
      if (relu) v = v > 0.f ? v : 0.f;
      C[(size_t)row * ldc + col] = v;
    }
  }
}

DEVINL float wave_sum(float v) {
#pragma unroll
  for (int o = 16; o > 0; o >>= 1) v += __shfl_xor(v, o, 32);
  return v;
}

// ---------------- kernels ----------------
__global__ void f32_to_f16_kernel(const float* __restrict__ src,
                                  _Float16* __restrict__ dst, int n) {
  int i = blockIdx.x * blockDim.x + threadIdx.x;
  if (i < n) dst[i] = (_Float16)src[i];
}

// Generic C[M,N] = A[M,K] @ W[N,K]^T + bias, optional ReLU. One wave per 32x32 tile.
__global__ __launch_bounds__(256) void gemm_xwt_kernel(
    const _Float16* __restrict__ A, const _Float16* __restrict__ Wm,
    const float* __restrict__ bias, float* __restrict__ C,
    int M, int N, int K, int relu) {
  int wave   = (blockIdx.x << 3) + (threadIdx.x >> 5);
  int tilesN = N >> 5;
  int tilesM = (M + 31) >> 5;
  if (wave >= tilesM * tilesN) return;  // wave-uniform: EXEC stays all-ones
  int m0 = (wave / tilesN) << 5;
  int n0 = (wave % tilesN) << 5;
  v8f a00 = {}, a01 = {}, a10 = {}, a11 = {};
  for (int k0 = 0; k0 < K; k0 += 32) {
    v16h A0 = load_a_frag(A, K, m0,      k0, M - 1);
    v16h A1 = load_a_frag(A, K, m0 + 16, k0, M - 1);
    v16h B0 = load_b_frag(Wm, K, n0,      k0);
    v16h B1 = load_b_frag(Wm, K, n0 + 16, k0);
    a00 = wmma16x16x32(A0, B0, a00);
    a01 = wmma16x16x32(A0, B1, a01);
    a10 = wmma16x16x32(A1, B0, a10);
    a11 = wmma16x16x32(A1, B1, a11);
  }
  store_c_f32(C, N, M, m0,      n0,      a00, bias, relu);
  store_c_f32(C, N, M, m0,      n0 + 16, a01, bias, relu);
  store_c_f32(C, N, M, m0 + 16, n0,      a10, bias, relu);
  store_c_f32(C, N, M, m0 + 16, n0 + 16, a11, bias, relu);
}

// Attention: one wave per (window, head). WS=5 tokens, HD=64 -> 2 comps/lane.
__global__ __launch_bounds__(256) void attn_kernel(
    const float* __restrict__ qkv, _Float16* __restrict__ o) {
  int gw = (blockIdx.x << 3) + (threadIdx.x >> 5);
  if (gw >= NWIN * H_) return;
  int lane = threadIdx.x & 31;
  int win  = gw >> 2;
  int h    = gw & (H_ - 1);
  int b    = win / W_;
  int w    = win - b * W_;
  float q[WS_][2], k[WS_][2], v[WS_][2];
#pragma unroll
  for (int t = 0; t < WS_; ++t) {
    const float* p = qkv + (size_t)(b * S_ + w + t) * (3 * E_) + h * HD_ + lane * 2;
    q[t][0] = p[0];          q[t][1] = p[1];
    k[t][0] = p[E_];         k[t][1] = p[E_ + 1];
    v[t][0] = p[2 * E_];     v[t][1] = p[2 * E_ + 1];
  }
  float sc[WS_][WS_];
#pragma unroll
  for (int t = 0; t < WS_; ++t)
#pragma unroll
    for (int u = 0; u < WS_; ++u) {
      float d = q[t][0] * k[u][0] + q[t][1] * k[u][1];
      sc[t][u] = wave_sum(d) * 0.125f;  // 1/sqrt(64)
    }
#pragma unroll
  for (int t = 0; t < WS_; ++t) {
    float mx = sc[t][0];
#pragma unroll
    for (int u = 1; u < WS_; ++u) mx = fmaxf(mx, sc[t][u]);
    float s = 0.f;
#pragma unroll
    for (int u = 0; u < WS_; ++u) { sc[t][u] = __expf(sc[t][u] - mx); s += sc[t][u]; }
    float inv = 1.f / s;
    float o0 = 0.f, o1 = 0.f;
#pragma unroll
    for (int u = 0; u < WS_; ++u) { o0 += sc[t][u] * v[u][0]; o1 += sc[t][u] * v[u][1]; }
    _Float16* op = o + (size_t)(win * WS_ + t) * E_ + h * HD_ + lane * 2;
    op[0] = (_Float16)(o0 * inv);
    op[1] = (_Float16)(o1 * inv);
  }
}

// LN1: x1 = LN(window_x + o_proj). One wave per row; writes f32 (residual) + f16 (GEMM A).
__global__ __launch_bounds__(256) void ln1_kernel(
    const float* __restrict__ hidden, const float* __restrict__ oproj,
    const float* __restrict__ g, const float* __restrict__ bb,
    float* __restrict__ x1f, _Float16* __restrict__ x1h) {
  int row = (blockIdx.x << 3) + (threadIdx.x >> 5);
  if (row >= NROW) return;
  int lane = threadIdx.x & 31;
  int win = row / WS_, t = row - win * WS_;
  int b = win / W_, w = win - b * W_;
  const float* hr = hidden + (size_t)(b * S_ + w + t) * E_;
  const float* orw = oproj + (size_t)row * E_;
  float vals[8], s = 0.f, s2 = 0.f;
#pragma unroll
  for (int j = 0; j < 8; ++j) {
    int c = lane + (j << 5);
    float v = hr[c] + orw[c];
    vals[j] = v; s += v; s2 += v * v;
  }
  s  = wave_sum(s);
  s2 = wave_sum(s2);
  float mean = s * (1.f / E_);
  float var  = s2 * (1.f / E_) - mean * mean;
  float rs   = rsqrtf(var + EPS_);
#pragma unroll
  for (int j = 0; j < 8; ++j) {
    int c = lane + (j << 5);
    float y = (vals[j] - mean) * rs * g[c] + bb[c];
    x1f[(size_t)row * E_ + c] = y;
    x1h[(size_t)row * E_ + c] = (_Float16)y;
  }
}

// Fused FFN v2: 64 rows/WG, 512 threads (16 waves).
//   phase 1: h[64][2048] = relu(x1 @ W1^T + b1) -> LDS (f16, padded rows)
//   phase 2: out[64][256] = h @ W2^T + b2 + x1 residual (accs in registers)
//   phase 3: xout overlays hbuf (h is dead after barrier), LN2 -> x2 f16
// LDS = 64*(2048+16)*2 = 258KB <= 320KB WGP LDS. Halves lin1/lin2 L2 traffic
// vs the 32-row version (weights amortized over twice the rows).
__global__ __launch_bounds__(512) void ffn_fused_kernel(
    const _Float16* __restrict__ x1h, const float* __restrict__ x1f,
    const _Float16* __restrict__ w1, const float* __restrict__ b1,
    const _Float16* __restrict__ w2, const float* __restrict__ b2,
    const float* __restrict__ g2, const float* __restrict__ bln2,
    _Float16* __restrict__ x2h) {
  extern __shared__ char smem[];
  constexpr int HLD = FF_ + 16;   // pad to stagger LDS banks
  constexpr int XLD = E_ + 8;
  _Float16* hbuf = (_Float16*)smem;      // [64][HLD] f16
  float*    xout = (float*)smem;         // [64][XLD] f32, overlays hbuf (phase 3)
  int wv   = threadIdx.x >> 5;           // 0..15
  int lane = threadIdx.x & 31;
  int m0   = blockIdx.x << 6;            // 64-row tile base (last WG partial)

  // ---- phase 1: 128 tiles of 32x32 over h[64][2048]; 8 tiles per wave ----
#pragma unroll 1
  for (int i = 0; i < 8; ++i) {
    int tile = (wv << 3) + i;            // 0..127
    int mi   = tile >> 6;                // 0..1  (32-row half)
    int nj   = tile & 63;                // 0..63 (32-col slice of FF)
    int mloc = mi << 5;
    int n0   = nj << 5;
    v8f a00 = {}, a01 = {}, a10 = {}, a11 = {};
    for (int k0 = 0; k0 < E_; k0 += 32) {
      v16h A0 = load_a_frag(x1h, E_, m0 + mloc,      k0, NROW - 1);
      v16h A1 = load_a_frag(x1h, E_, m0 + mloc + 16, k0, NROW - 1);
      v16h B0 = load_b_frag(w1, E_, n0,      k0);
      v16h B1 = load_b_frag(w1, E_, n0 + 16, k0);
      a00 = wmma16x16x32(A0, B0, a00);
      a01 = wmma16x16x32(A0, B1, a01);
      a10 = wmma16x16x32(A1, B0, a10);
      a11 = wmma16x16x32(A1, B1, a11);
    }
#pragma unroll
    for (int sm = 0; sm < 2; ++sm)
#pragma unroll
      for (int sn = 0; sn < 2; ++sn) {
        v8f acc = (sm == 0) ? (sn == 0 ? a00 : a01) : (sn == 0 ? a10 : a11);
        int col = n0 + (sn << 4) + (lane & 15);
        int rb  = mloc + (sm << 4) + ((lane >> 4) << 3);
        float bv = b1[col];
#pragma unroll
        for (int r = 0; r < 8; ++r) {
          float v = acc[r] + bv;
          hbuf[(rb + r) * HLD + col] = (_Float16)(v > 0.f ? v : 0.f);
        }
      }
  }
  __syncthreads();

  // ---- phase 2: out[64][256] = h @ W2^T; 16 tiles of 32x32, one per wave ----
  int mi2 = wv >> 3;                     // 0..1
  int nj2 = wv & 7;                      // 0..7
  int ml2 = mi2 << 5;
  int n02 = nj2 << 5;
  v8f c00 = {}, c01 = {}, c10 = {}, c11 = {};
  for (int k0 = 0; k0 < FF_; k0 += 32) {
    v16h A0 = load_a_frag(hbuf, HLD, ml2,      k0, 63);  // LDS -> ds_load
    v16h A1 = load_a_frag(hbuf, HLD, ml2 + 16, k0, 63);
    v16h B0 = load_b_frag(w2, FF_, n02,      k0);
    v16h B1 = load_b_frag(w2, FF_, n02 + 16, k0);
    c00 = wmma16x16x32(A0, B0, c00);
    c01 = wmma16x16x32(A0, B1, c01);
    c10 = wmma16x16x32(A1, B0, c10);
    c11 = wmma16x16x32(A1, B1, c11);
  }
  __syncthreads();   // all waves done reading h; safe to overlay with xout

#pragma unroll
  for (int sm = 0; sm < 2; ++sm)
#pragma unroll
    for (int sn = 0; sn < 2; ++sn) {
      v8f acc = (sm == 0) ? (sn == 0 ? c00 : c01) : (sn == 0 ? c10 : c11);
      int col = n02 + (sn << 4) + (lane & 15);
      int rb  = ml2 + (sm << 4) + ((lane >> 4) << 3);
      float bv = b2[col];
#pragma unroll
      for (int r = 0; r < 8; ++r) {
        int rl = rb + r;
        int grow = m0 + rl;
        if (grow > NROW - 1) grow = NROW - 1;  // clamp residual read (tail WG)
        xout[rl * XLD + col] = acc[r] + bv + x1f[(size_t)grow * E_ + col];
      }
    }
  __syncthreads();

  // ---- phase 3: LN2 per row (16 waves x 4 rows), write x2 f16 ----
#pragma unroll 1
  for (int rr = 0; rr < 4; ++rr) {
    int rl  = (wv << 2) + rr;
    int row = m0 + rl;
    float vals[8], s = 0.f, s2 = 0.f;
#pragma unroll
    for (int j = 0; j < 8; ++j) {
      int c = lane + (j << 5);
      float v = xout[rl * XLD + c];
      vals[j] = v; s += v; s2 += v * v;
    }
    s  = wave_sum(s);
    s2 = wave_sum(s2);
    float mean = s * (1.f / E_);
    float var  = s2 * (1.f / E_) - mean * mean;
    float rs   = rsqrtf(var + EPS_);
    if (row < NROW) {
#pragma unroll
      for (int j = 0; j < 8; ++j) {
        int c = lane + (j << 5);
        x2h[(size_t)row * E_ + c] =
            (_Float16)((vals[j] - mean) * rs * g2[c] + bln2[c]);
      }
    }
  }
}

// Max-pool over the 5 tokens of each window.
__global__ void pool_kernel(const _Float16* __restrict__ x2h,
                            _Float16* __restrict__ pooled) {
  int i = blockIdx.x * blockDim.x + threadIdx.x;
  if (i >= NWIN * E_) return;
  int win = i >> 8, c = i & 255;
  float m = -3.4e38f;
#pragma unroll
  for (int t = 0; t < WS_; ++t) {
    float v = (float)x2h[((size_t)(win * WS_ + t) << 8) + c];
    m = fmaxf(m, v);
  }
  pooled[i] = (_Float16)m;
}

// window_num / seq_lens tail outputs (stored as float in the flat output).
__global__ void tail_kernel(const int* __restrict__ seq, float* __restrict__ out) {
  int i = threadIdx.x;
  if (i < B_) {
    int wn = seq[i] - (WS_ - 1);
    if (wn < 1) wn = 1;
    out[i]      = (float)wn;
    out[B_ + i] = (float)seq[i];
  }
}

// ---------------- workspace layout (bytes), lifetime-aliased ----------------
constexpr size_t SZ_HID16 = (size_t)NTOK * E_ * 2;        //  8.39 MB
constexpr size_t SZ_WIN   = (size_t)3 * E_ * E_ * 2;      //  0.39 MB
constexpr size_t SZ_WOUT  = (size_t)E_ * E_ * 2;          //  0.13 MB
constexpr size_t SZ_WL1   = (size_t)FF_ * E_ * 2;         //  1.05 MB
constexpr size_t SZ_WL2   = (size_t)E_ * FF_ * 2;         //  1.05 MB
constexpr size_t SZ_WDEC  = (size_t)NI_ * E_ * 2;         //  0.03 MB
constexpr size_t SZ_POOL  = (size_t)NWIN * E_ * 2;        //  8.37 MB
constexpr size_t SZ_QKV   = (size_t)NTOK * 3 * E_ * 4;    // 50.33 MB
constexpr size_t SZ_ATTO  = (size_t)NROW * E_ * 2;        // 41.86 MB
constexpr size_t SZ_OPROJ = (size_t)NROW * E_ * 4;        // 83.72 MB
constexpr size_t SZ_X1F16 = (size_t)NROW * E_ * 2;        // 41.86 MB

constexpr size_t OFF_HID16 = 0;
constexpr size_t OFF_WIN   = OFF_HID16 + SZ_HID16;
constexpr size_t OFF_WOUT  = OFF_WIN   + SZ_WIN;
constexpr size_t OFF_WL1   = OFF_WOUT  + SZ_WOUT;
constexpr size_t OFF_WL2   = OFF_WL1   + SZ_WL1;
constexpr size_t OFF_WDEC  = OFF_WL2   + SZ_WL2;
constexpr size_t OFF_POOL  = OFF_WDEC  + SZ_WDEC;
constexpr size_t OFF_QKV   = OFF_POOL  + SZ_POOL;          // region B (qkv+attn_o)
constexpr size_t OFF_ATTO  = OFF_QKV   + SZ_QKV;
constexpr size_t OFF_OPROJ = OFF_ATTO  + SZ_ATTO;          // region C
constexpr size_t OFF_X1F16 = OFF_OPROJ + SZ_OPROJ;
// aliases (dead-by-then regions): x1_f32 over [QKV..ATTO), x2_f16 over OPROJ
constexpr size_t OFF_X1F32 = OFF_QKV;     // needs 83.72MB <= 92.19MB available
constexpr size_t OFF_X2F16 = OFF_OPROJ;   // needs 41.86MB <= 83.72MB available

extern "C" void kernel_launch(void* const* d_in, const int* in_sizes, int n_in,
                              void* d_out, int out_size, void* d_ws, size_t ws_size,
                              hipStream_t stream) {
  const float* hidden  = (const float*)d_in[0];
  const int*   seqlens = (const int*)  d_in[1];
  const float* in_w    = (const float*)d_in[2];
  const float* in_b    = (const float*)d_in[3];
  const float* out_w   = (const float*)d_in[4];
  const float* out_b   = (const float*)d_in[5];
  const float* ln1_g   = (const float*)d_in[6];
  const float* ln1_b   = (const float*)d_in[7];
  const float* l1_w    = (const float*)d_in[8];
  const float* l1_b    = (const float*)d_in[9];
  const float* l2_w    = (const float*)d_in[10];
  const float* l2_b    = (const float*)d_in[11];
  const float* ln2_g   = (const float*)d_in[12];
  const float* ln2_b   = (const float*)d_in[13];
  const float* dec_w   = (const float*)d_in[14];
  const float* dec_b   = (const float*)d_in[15];
  float* out = (float*)d_out;

  char* ws = (char*)d_ws;
  _Float16* hid16 = (_Float16*)(ws + OFF_HID16);
  _Float16* w_in  = (_Float16*)(ws + OFF_WIN);
  _Float16* w_out = (_Float16*)(ws + OFF_WOUT);
  _Float16* w_l1  = (_Float16*)(ws + OFF_WL1);
  _Float16* w_l2  = (_Float16*)(ws + OFF_WL2);
  _Float16* w_dec = (_Float16*)(ws + OFF_WDEC);
  _Float16* pooled= (_Float16*)(ws + OFF_POOL);
  float*    qkv   = (float*)   (ws + OFF_QKV);
  _Float16* attn_o= (_Float16*)(ws + OFF_ATTO);
  float*    oproj = (float*)   (ws + OFF_OPROJ);
  float*    x1f   = (float*)   (ws + OFF_X1F32);
  _Float16* x1h   = (_Float16*)(ws + OFF_X1F16);
  _Float16* x2h   = (_Float16*)(ws + OFF_X2F16);

  auto cvt = [&](const float* s, _Float16* d, int n) {
    f32_to_f16_kernel<<<(n + 255) / 256, 256, 0, stream>>>(s, d, n);
  };
  // 1) f16 copies of activations/weights
  cvt(hidden, hid16, NTOK * E_);
  cvt(in_w,  w_in,  3 * E_ * E_);
  cvt(out_w, w_out, E_ * E_);
  cvt(l1_w,  w_l1,  FF_ * E_);
  cvt(l2_w,  w_l2,  E_ * FF_);
  cvt(dec_w, w_dec, NI_ * E_);

  auto gemm = [&](const _Float16* A, const _Float16* Wm, const float* bias,
                  float* C, int M, int N, int K, int relu) {
    int waves  = (((M + 31) >> 5)) * (N >> 5);
    int blocks = (waves + 7) >> 3;
    gemm_xwt_kernel<<<blocks, 256, 0, stream>>>(A, Wm, bias, C, M, N, K, relu);
  };

  // 2) per-token QKV projection (windows are shifted views -> hoist out of windows)
  gemm(hid16, w_in, in_b, qkv, NTOK, 3 * E_, E_, 0);

  // 3) windowed attention -> o rows (f16)
  attn_kernel<<<(NWIN * H_ + 7) / 8, 256, 0, stream>>>(qkv, attn_o);

  // 4) out-proj GEMM + residual/LN1
  gemm(attn_o, w_out, out_b, oproj, NROW, E_, E_, 0);
  ln1_kernel<<<(NROW + 7) / 8, 256, 0, stream>>>(hidden, oproj, ln1_g, ln1_b,
                                                 x1f, x1h);

  // 5) fused FFN + residual + LN2 (h stays in LDS; 64-row tiles, 16 waves)
  {
    constexpr size_t smem = (size_t)64 * (FF_ + 16) * sizeof(_Float16); // 258KB
    ffn_fused_kernel<<<(NROW + 63) / 64, 512, smem, stream>>>(
        x1h, x1f, w_l1, l1_b, w_l2, l2_b, ln2_g, ln2_b, x2h);
  }

  // 6) max-pool over window tokens, decode GEMM straight into d_out, tail ints
  pool_kernel<<<(NWIN * E_ + 255) / 256, 256, 0, stream>>>(x2h, pooled);
  gemm(pooled, w_dec, dec_b, out, NWIN, NI_, E_, 0);
  tail_kernel<<<1, 32, 0, stream>>>(seqlens, out + (size_t)NWIN * NI_);
}
